// AnalogicalReasoning_44796508897483
// MI455X (gfx1250) — compile-verified, matrix operand order checked by reading, not run on hardware
//
#include <hip/hip_runtime.h>
#include <hip/hip_bf16.h>

typedef __attribute__((ext_vector_type(16))) _Float16 v16h;
typedef __attribute__((ext_vector_type(8)))  _Float16 v8h;
typedef __attribute__((ext_vector_type(8)))  float    v8f;

static constexpr int kH  = 256;   // hidden / E
static constexpr float kEPS = 1e-5f;

__device__ __forceinline__ float waveSum(float v) {
#pragma unroll
  for (int off = 16; off > 0; off >>= 1) v += __shfl_xor(v, off, 32);
  return v;
}

// ---------------------------------------------------------------------------
// Wt[n*256 + k] = (f16) W[k*256 + n]   (one 256x256 matrix; grid=256, blk=256)
// ---------------------------------------------------------------------------
__global__ __launch_bounds__(256) void conv_tr_f16(
    const float* __restrict__ W, _Float16* __restrict__ Wt)
{
  const int idx = blockIdx.x * 256 + threadIdx.x;  // 0..65535
  const int k = idx >> 8;
  const int n = idx & 255;
  Wt[n * 256 + k] = (_Float16)W[idx];
}

// ---------------------------------------------------------------------------
// Plain f32 -> f16 elementwise convert (row-major preserved).
// ---------------------------------------------------------------------------
__global__ __launch_bounds__(256) void conv_f16(
    const float* __restrict__ X, _Float16* __restrict__ Y)
{
  const int idx = blockIdx.x * 256 + threadIdx.x;
  Y[idx] = (_Float16)X[idx];
}

// ---------------------------------------------------------------------------
// C[M,256] = A16[M,256] @ W[256,256] (+ bias[col]) (masked by mask[row] > 0)
// A16: (M,K) row-major f16.  Wt: W transposed to (N,K) row-major f16.
// One wave computes a 16x64 strip: 4 accumulators, A fragment reused 4x,
// 32 WMMAs/wave; inner loop = 2 A b128 loads + 8 B b128 loads + 4 WMMAs.
// TOut = float (f32 store) or _Float16 (feeds the next GEMM's A operand).
// ---------------------------------------------------------------------------
template <typename TOut>
__global__ __launch_bounds__(256) void gemm256(
    const _Float16* __restrict__ A, const _Float16* __restrict__ Wt,
    const float* __restrict__ bias, const float* __restrict__ mask,
    TOut* __restrict__ C, int M)
{
  const int tid  = threadIdx.x;
  const int lane = tid & 31;
  const int wv   = tid >> 5;
  const int half = lane >> 4;       // 0: lanes 0-15, 1: lanes 16-31
  const int l16  = lane & 15;

  const int grp = blockIdx.x * 8 + wv;   // 16-row strip * 64-col group
  const int tm  = grp >> 2;
  const int gn  = grp & 3;
  if (tm * 16 >= M) return;              // wave-uniform; EXEC stays all-1s

  const _Float16* arow = A + (size_t)(tm * 16 + l16) * kH;
  const int col0 = gn * 64 + l16;

  v8f acc0 = {}, acc1 = {}, acc2 = {}, acc3 = {};
#pragma unroll
  for (int kk = 0; kk < kH; kk += 32) {
    const int klo = kk + half * 8;       // K 0..7  (lanes<16) / 8..15
    const int khi = kk + 16 + half * 8;  // K 16..23 / 24..31

    const v8h alo = *(const v8h*)(arow + klo);   // global_load_b128
    const v8h ahi = *(const v8h*)(arow + khi);   // global_load_b128
    v16h af;
#pragma unroll
    for (int e = 0; e < 8; ++e) { af[e] = alo[e]; af[e + 8] = ahi[e]; }

#pragma unroll
    for (int t = 0; t < 4; ++t) {
      const _Float16* wrow = Wt + (size_t)(col0 + t * 16) * kH;
      const v8h blo = *(const v8h*)(wrow + klo);
      const v8h bhi = *(const v8h*)(wrow + khi);
      v16h bf;
#pragma unroll
      for (int e = 0; e < 8; ++e) { bf[e] = blo[e]; bf[e + 8] = bhi[e]; }
      const v8f c = (t == 0) ? acc0 : (t == 1) ? acc1 : (t == 2) ? acc2 : acc3;
      const v8f d = __builtin_amdgcn_wmma_f32_16x16x32_f16(
          false, af, false, bf, (short)0, c, false, false);
      if (t == 0) acc0 = d; else if (t == 1) acc1 = d;
      else if (t == 2) acc2 = d; else acc3 = d;
    }
  }

#pragma unroll
  for (int t = 0; t < 4; ++t) {
    const int col = col0 + t * 16;
    const float bv = bias ? bias[col] : 0.0f;
    const v8f acc = (t == 0) ? acc0 : (t == 1) ? acc1 : (t == 2) ? acc2 : acc3;
#pragma unroll
    for (int r = 0; r < 8; ++r) {
      const int row = tm * 16 + r + 8 * half;
      float v = acc[r] + bv;
      if (mask) v = (mask[row] > 0.0f) ? v : 0.0f;
      C[(size_t)row * kH + col] = (TOut)v;
    }
  }
}

// ---------------------------------------------------------------------------
// Hout16[row=b*256+i*16+j, :] = (f16) relu(LN(P[b*16+i,:] + Q[b*16+j,:]))
// (rel_b1 already folded into P via GEMM bias). One wave per row.
// Output f16: consumed only as the next GEMM's A operand.
// ---------------------------------------------------------------------------
__global__ __launch_bounds__(256) void pair_ln_relu(
    const float* __restrict__ P, const float* __restrict__ Q,
    const float* __restrict__ g, const float* __restrict__ be,
    _Float16* __restrict__ Hout)
{
  const int lane = threadIdx.x & 31;
  const int wv   = threadIdx.x >> 5;
  const int row  = blockIdx.x * 8 + wv;   // 0..2047
  const int b    = row >> 8;
  const int rem  = row & 255;
  const int i    = rem >> 4;
  const int j    = rem & 15;
  const float* prow = P + (size_t)(b * 16 + i) * kH;
  const float* qrow = Q + (size_t)(b * 16 + j) * kH;

  float v[8];
  float s = 0.f;
#pragma unroll
  for (int k = 0; k < 8; ++k) {
    const int idx = lane + 32 * k;
    v[k] = prow[idx] + qrow[idx];
    s += v[k];
  }
  const float m = waveSum(s) * (1.0f / kH);
  float s2 = 0.f;
#pragma unroll
  for (int k = 0; k < 8; ++k) { const float d = v[k] - m; s2 += d * d; }
  const float rs = rsqrtf(waveSum(s2) * (1.0f / kH) + kEPS);

  _Float16* orow = Hout + (size_t)row * kH;
#pragma unroll
  for (int k = 0; k < 8; ++k) {
    const int idx = lane + 32 * k;
    const float y = (v[k] - m) * rs * g[idx] + be[idx];
    orow[idx] = (_Float16)fmaxf(y, 0.0f);
  }
}

// ---------------------------------------------------------------------------
// Mapping scores + fused softmax.
// Block = one (b,s) row: 8 waves loop over t; per pair: LN+relu+dot (wave32).
// ---------------------------------------------------------------------------
__global__ __launch_bounds__(256) void map_score_softmax(
    const float* __restrict__ HA, const float* __restrict__ HB,
    const float* __restrict__ g, const float* __restrict__ be,
    const float* __restrict__ w2, const float* __restrict__ b2,
    float* __restrict__ out)
{
  __shared__ float sHa[kH], sG[kH], sBe[kH], sW2[kH], sLog[kH], sRed[kH];
  const int tid  = threadIdx.x;
  const int lane = tid & 31;
  const int wv   = tid >> 5;
  const int bs   = blockIdx.x;          // b*256 + s
  const int b    = bs >> 8;

  sHa[tid] = HA[(size_t)bs * kH + tid]; // includes map_b1
  sG[tid]  = g[tid];
  sBe[tid] = be[tid];
  sW2[tid] = w2[tid];
  __syncthreads();
  const float bias2 = b2[0];

  for (int t = wv; t < 256; t += 8) {
    const float* hb = HB + (size_t)(b * 256 + t) * kH;
    float v[8];
    float s = 0.f;
#pragma unroll
    for (int k = 0; k < 8; ++k) {
      const int idx = lane + 32 * k;    // coalesced, LDS bank-conflict-free
      v[k] = sHa[idx] + hb[idx];
      s += v[k];
    }
    const float m = waveSum(s) * (1.0f / kH);
    float s2 = 0.f;
#pragma unroll
    for (int k = 0; k < 8; ++k) { const float d = v[k] - m; s2 += d * d; }
    const float rs = rsqrtf(waveSum(s2) * (1.0f / kH) + kEPS);
    float dot = 0.f;
#pragma unroll
    for (int k = 0; k < 8; ++k) {
      const int idx = lane + 32 * k;
      const float y = (v[k] - m) * rs * sG[idx] + sBe[idx];
      dot += fmaxf(y, 0.f) * sW2[idx];
    }
    dot = waveSum(dot);
    if (lane == 0) sLog[t] = dot + bias2;
  }
  __syncthreads();

  // block softmax over 256 logits
  const float l = sLog[tid];
  sRed[tid] = l;
  __syncthreads();
  for (int st = 128; st > 0; st >>= 1) {
    if (tid < st) sRed[tid] = fmaxf(sRed[tid], sRed[tid + st]);
    __syncthreads();
  }
  const float mx = sRed[0];
  __syncthreads();
  const float e = __expf(l - mx);
  sRed[tid] = e;
  __syncthreads();
  for (int st = 128; st > 0; st >>= 1) {
    if (tid < st) sRed[tid] += sRed[tid + st];
    __syncthreads();
  }
  out[(size_t)bs * 256 + tid] = e / sRed[0];
}

// ---------------------------------------------------------------------------
// Correspondence scores (no LN) + fused softmax over 16 targets.
// ---------------------------------------------------------------------------
__global__ __launch_bounds__(256) void cor_score_softmax(
    const float* __restrict__ HA, const float* __restrict__ HB,
    const float* __restrict__ w2, const float* __restrict__ b2,
    float* __restrict__ out)
{
  __shared__ float sHa[kH], sW2[kH], sLog[16];
  const int tid  = threadIdx.x;
  const int lane = tid & 31;
  const int wv   = tid >> 5;
  const int bs   = blockIdx.x;          // b*16 + s
  const int b    = bs >> 4;

  sHa[tid] = HA[(size_t)bs * kH + tid]; // includes cor_b1
  sW2[tid] = w2[tid];
  __syncthreads();
  const float bias2 = b2[0];

  for (int t = wv; t < 16; t += 8) {
    const float* hb = HB + (size_t)(b * 16 + t) * kH;
    float dot = 0.f;
#pragma unroll
    for (int k = 0; k < 8; ++k) {
      const int idx = lane + 32 * k;
      dot += fmaxf(sHa[idx] + hb[idx], 0.f) * sW2[idx];
    }
    dot = waveSum(dot);
    if (lane == 0) sLog[t] = dot + bias2;
  }
  __syncthreads();

  if (tid < 16) {
    float mx = -3.4e38f;
    for (int k = 0; k < 16; ++k) mx = fmaxf(mx, sLog[k]);
    float sum = 0.f;
    for (int k = 0; k < 16; ++k) sum += __expf(sLog[k] - mx);
    out[(size_t)bs * 16 + tid] = __expf(sLog[tid] - mx) / sum;
  }
}

// ---------------------------------------------------------------------------
extern "C" void kernel_launch(void* const* d_in, const int* in_sizes, int n_in,
                              void* d_out, int out_size, void* d_ws, size_t ws_size,
                              hipStream_t stream) {
  (void)in_sizes; (void)n_in; (void)out_size; (void)ws_size;

  const float* src_ent  = (const float*)d_in[0];   // (8,16,256)
  const float* src_rels = (const float*)d_in[1];   // (8,16,16)
  const float* tgt_ent  = (const float*)d_in[2];
  const float* tgt_rels = (const float*)d_in[3];
  const float* rel_W1   = (const float*)d_in[4];   // (512,256)
  const float* rel_b1   = (const float*)d_in[5];
  const float* rel_g1   = (const float*)d_in[6];
  const float* rel_be1  = (const float*)d_in[7];
  const float* rel_W2   = (const float*)d_in[8];   // (256,256)
  const float* rel_b2   = (const float*)d_in[9];
  const float* map_W1   = (const float*)d_in[10];  // (512,256)
  const float* map_b1   = (const float*)d_in[11];
  const float* map_g    = (const float*)d_in[12];
  const float* map_be   = (const float*)d_in[13];
  const float* map_W2   = (const float*)d_in[14];  // (256,1)
  const float* map_b2   = (const float*)d_in[15];
  const float* cor_W1   = (const float*)d_in[16];  // (512,256)
  const float* cor_b1   = (const float*)d_in[17];
  const float* cor_W2   = (const float*)d_in[18];  // (256,1)
  const float* cor_b2   = (const float*)d_in[19];

  // ---- f32 workspace -------------------------------------------------------
  float* ws    = (float*)d_ws;
  float* srcP  = ws;               // 128*256   ent @ Wi + rel_b1
  float* srcQ  = srcP + 32768;     // 128*256   ent @ Wj
  float* tgtP  = srcQ + 32768;
  float* tgtQ  = tgtP + 32768;
  float* mapHa = tgtQ + 32768;     // 2048*256  sRel @ mapWa + map_b1
  float* mapHb = mapHa + 524288;   // 2048*256  tRel @ mapWb
  float* cHa   = mapHb + 524288;   // 128*256
  float* cHb   = cHa + 32768;      // 128*256

  // ---- f16 workspace (16B-aligned: offsets are multiples of 8 floats) ------
  _Float16* h16   = (_Float16*)(cHb + 32768);
  _Float16* entS  = h16;                 // 128*256   f16 source entities
  _Float16* entT  = entS + 32768;        // 128*256   f16 target entities
  _Float16* Hs16  = entT + 32768;        // 2048*256  relu(LN(..)) source
  _Float16* Ht16  = Hs16 + 524288;       // 2048*256  relu(LN(..)) target
  _Float16* sRel  = Ht16 + 524288;       // 2048*256  masked src relations
  _Float16* tRel  = sRel + 524288;       // 2048*256  masked tgt relations
  _Float16* wtRelI = tRel + 524288;      // 7 x 256x256 f16 transposed weights
  _Float16* wtRelJ = wtRelI + 65536;
  _Float16* wtRel2 = wtRelJ + 65536;
  _Float16* wtMapA = wtRel2 + 65536;
  _Float16* wtMapB = wtMapA + 65536;
  _Float16* wtCorA = wtMapB + 65536;
  _Float16* wtCorB = wtCorA + 65536;

  float* outMap = (float*)d_out;            // (8,256,256)
  float* outCor = outMap + 8 * 256 * 256;   // (8,16,16)

  const dim3 blk(256);

  // Stage 0: one-time weight transpose+convert, entity convert
  conv_tr_f16<<<256, blk, 0, stream>>>(rel_W1,            wtRelI);
  conv_tr_f16<<<256, blk, 0, stream>>>(rel_W1 + 256 * kH, wtRelJ);
  conv_tr_f16<<<256, blk, 0, stream>>>(rel_W2,            wtRel2);
  conv_tr_f16<<<256, blk, 0, stream>>>(map_W1,            wtMapA);
  conv_tr_f16<<<256, blk, 0, stream>>>(map_W1 + 256 * kH, wtMapB);
  conv_tr_f16<<<256, blk, 0, stream>>>(cor_W1,            wtCorA);
  conv_tr_f16<<<256, blk, 0, stream>>>(cor_W1 + 256 * kH, wtCorB);
  conv_f16<<<128, blk, 0, stream>>>(src_ent, entS);
  conv_f16<<<128, blk, 0, stream>>>(tgt_ent, entT);

  // Stage A: entity projections (M=128 -> 32 wave-groups -> 4 blocks)
  gemm256<float><<<4, blk, 0, stream>>>(entS, wtRelI, rel_b1, nullptr, srcP, 128);
  gemm256<float><<<4, blk, 0, stream>>>(entS, wtRelJ, nullptr, nullptr, srcQ, 128);
  gemm256<float><<<4, blk, 0, stream>>>(entT, wtRelI, rel_b1, nullptr, tgtP, 128);
  gemm256<float><<<4, blk, 0, stream>>>(entT, wtRelJ, nullptr, nullptr, tgtQ, 128);

  // Stage B1: pairwise sum + LN + relu -> f16 (2048,256)
  pair_ln_relu<<<256, blk, 0, stream>>>(srcP, srcQ, rel_g1, rel_be1, Hs16);
  pair_ln_relu<<<256, blk, 0, stream>>>(tgtP, tgtQ, rel_g1, rel_be1, Ht16);

  // Stage B2: relation encodings with (rel>0) mask -> f16 (M=2048 -> 64 blocks)
  gemm256<_Float16><<<64, blk, 0, stream>>>(Hs16, wtRel2, rel_b2, src_rels, sRel, 2048);
  gemm256<_Float16><<<64, blk, 0, stream>>>(Ht16, wtRel2, rel_b2, tgt_rels, tRel, 2048);

  // Stage C: mapping projections -> f32
  gemm256<float><<<64, blk, 0, stream>>>(sRel, wtMapA, map_b1, nullptr, mapHa, 2048);
  gemm256<float><<<64, blk, 0, stream>>>(tRel, wtMapB, nullptr, nullptr, mapHb, 2048);

  // Stage C2: LN'd pairwise scores + softmax -> d_out mapping part
  map_score_softmax<<<2048, blk, 0, stream>>>(mapHa, mapHb, map_g, map_be,
                                              map_W2, map_b2, outMap);

  // Stage D: correspondence path
  gemm256<float><<<4, blk, 0, stream>>>(entS, wtCorA, cor_b1, nullptr, cHa, 128);
  gemm256<float><<<4, blk, 0, stream>>>(entT, wtCorB, nullptr, nullptr, cHb, 128);
  cor_score_softmax<<<128, blk, 0, stream>>>(cHa, cHb, cor_W2, cor_b2, outCor);
}
